// MLAAttention_7765300871299
// MI455X (gfx1250) — matrix operand order repaired and checked by reading, not compiled, not run on hardware
//
#include <hip/hip_runtime.h>
#include <hip/hip_bf16.h>
#include <cstdint>

// ---------------------------------------------------------------------------
// MLA attention for MI455X (gfx1250, wave32, WMMA + async global->LDS)
//   q = x@Wq+bq ; lat = x@Wdown+bdown ; kv = lat@Wup+bup
//   per-token 16x16 head attention with RoPE ; out = att_out@Wout+bout
// GEMMs: bf16 inputs, v_wmma_f32_16x16x32_bf16, fp32 accumulate,
//        128x128 block tile, double-buffered global_load_async_to_lds_b128.
// Attention: v_wmma_f32_16x16x4_f32.
// ---------------------------------------------------------------------------

typedef __bf16  v16bf __attribute__((ext_vector_type(16)));
typedef float   v8f   __attribute__((ext_vector_type(8)));
typedef float   v2f   __attribute__((ext_vector_type(2)));

union AFrag16 { v16bf v; uint4 u[2]; };

// ---------------------- CDNA5 async global->LDS ----------------------------

__device__ __forceinline__ void async_load_b128(unsigned lds_off, const void* gaddr) {
  asm volatile("global_load_async_to_lds_b128 %0, %1, off"
               :: "v"(lds_off), "v"((unsigned long long)(uintptr_t)gaddr)
               : "memory");
}

__device__ __forceinline__ void wait_asynccnt0() {
  asm volatile("s_wait_asynccnt 0x0" ::: "memory");
}

// low 32 bits of the generic (flat) address of a shared object are the
// workgroup-relative LDS byte offset
__device__ __forceinline__ unsigned lds_off32(const void* p) {
  return (unsigned)(uintptr_t)p;
}

// ------------------------- conversion kernels ------------------------------

__global__ __launch_bounds__(256) void cvt_f32_to_bf16(const float* __restrict__ in,
                                                       __bf16* __restrict__ out, int n) {
  int i = blockIdx.x * 256 + threadIdx.x;
  if (i < n) out[i] = (__bf16)in[i];
}

// W: K x N (row-major fp32)  ->  Wt: N x K (row-major bf16)
__global__ __launch_bounds__(256) void cvt_transpose_bf16(const float* __restrict__ W,
                                                          __bf16* __restrict__ Wt,
                                                          int K, int N) {
  int i = blockIdx.x * 256 + threadIdx.x;
  if (i < K * N) {
    int n = i / K;
    int k = i - n * K;
    Wt[i] = (__bf16)W[(size_t)k * N + n];
  }
}

// ------------------------------ WMMA GEMM ----------------------------------
// C[M x N] = A[M x K] * Bt[N x K]^T + bias,  A/Bt bf16 row-major (K-major).
// Block tile 128x128, BK=32, 256 threads (8 waves).
// Wave (waveM = wid&3, waveN = wid>>2) owns a 32x64 sub-tile:
//   2 A fragments x 4 B fragments -> 8 accumulators, 8 WMMAs / K-step.
// Global->LDS staging via async b128 loads, double-buffered.

#define LDS_STRIDE 48                  // 32 K elems + 16 pad -> 96B rows, 16B aligned
#define TILE_ELEMS (128 * LDS_STRIDE)  // one 128x32 tile (padded)

template <bool OUT_BF16>
__global__ __launch_bounds__(256) void gemm_bf16_wmma(const __bf16* __restrict__ A,
                                                      const __bf16* __restrict__ Bt,
                                                      const float* __restrict__ bias,
                                                      void* __restrict__ Cout,
                                                      int M, int N, int K) {
  __shared__ __bf16 sA[2 * TILE_ELEMS];
  __shared__ __bf16 sB[2 * TILE_ELEMS];

  const int tid   = threadIdx.x;
  const int lane  = tid & 31;
  const int wid   = tid >> 5;
  const int waveM = wid & 3;       // 0..3  -> 32-row slab
  const int waveN = wid >> 2;      // 0..1  -> 64-col slab
  const int hi    = lane >> 4;
  const int nl    = lane & 15;

  const int blockN = (int)blockIdx.x * 128;
  const int blockM = (int)blockIdx.y * 128;

  const __bf16* gA0 = A  + (size_t)blockM * K;
  const __bf16* gB0 = Bt + (size_t)blockN * K;

  v8f c[2][4];
#pragma unroll
  for (int i = 0; i < 2; ++i)
#pragma unroll
    for (int j = 0; j < 4; ++j) c[i][j] = (v8f){};

  const int nsteps = K >> 5;

  // issue async staging for stage s into buffer (s&1): 2 A + 2 B chunks/thread
  auto stage_async = [&](int buf, int k0) {
    __bf16* dA = sA + buf * TILE_ELEMS;
    __bf16* dB = sB + buf * TILE_ELEMS;
#pragma unroll
    for (int j = 0; j < 2; ++j) {
      int chunk = tid + j * 256;       // 0..511 : 16B chunks of a 128x32 tile
      int row   = chunk >> 2;          // 0..127
      int c8    = (chunk & 3) * 8;     // 0,8,16,24
      async_load_b128(lds_off32(dA + row * LDS_STRIDE + c8),
                      gA0 + (size_t)row * K + k0 + c8);
      async_load_b128(lds_off32(dB + row * LDS_STRIDE + c8),
                      gB0 + (size_t)row * K + k0 + c8);
    }
  };

  stage_async(0, 0);

  for (int kt = 0; kt < nsteps; ++kt) {
    wait_asynccnt0();        // this wave's contribution to stage kt is resident
    __syncthreads();         // everyone's is; also fences buffer reuse
    if (kt + 1 < nsteps) stage_async((kt + 1) & 1, (kt + 1) * 32);

    const __bf16* bA = sA + (kt & 1) * TILE_ELEMS;
    const __bf16* bB = sB + (kt & 1) * TILE_ELEMS;

    // A fragments (16x32 bf16): lane -> M = row, K = kA..+7 and kA+16..+23
    const int kA = hi * 8;
    AFrag16 fa[2];
#pragma unroll
    for (int i = 0; i < 2; ++i) {
      const __bf16* p = bA + (waveM * 32 + i * 16 + nl) * LDS_STRIDE + kA;
      fa[i].u[0] = *(const uint4*)(p);
      fa[i].u[1] = *(const uint4*)(p + 16);
    }
    // B fragments (32x16 bf16): lane -> N = col, K = kB..+7 and kB+16..+23
    const int kB = hi * 8;
#pragma unroll
    for (int j = 0; j < 4; ++j) {
      AFrag16 fb;
      const __bf16* p = bB + (waveN * 64 + j * 16 + nl) * LDS_STRIDE + kB;
      fb.u[0] = *(const uint4*)(p);
      fb.u[1] = *(const uint4*)(p + 16);
      c[0][j] = __builtin_amdgcn_wmma_f32_16x16x32_bf16(false, fa[0].v, false, fb.v,
                                                        (short)0, c[0][j], false, false);
      c[1][j] = __builtin_amdgcn_wmma_f32_16x16x32_bf16(false, fa[1].v, false, fb.v,
                                                        (short)0, c[1][j], false, false);
    }
  }

  // epilogue: C layout — reg r: M = r + 8*hi, N = nl
#pragma unroll
  for (int i = 0; i < 2; ++i) {
    const int rowBase = blockM + waveM * 32 + i * 16 + 8 * hi;
#pragma unroll
    for (int j = 0; j < 4; ++j) {
      const int col = blockN + waveN * 64 + j * 16 + nl;
      const float b = bias[col];
#pragma unroll
      for (int r = 0; r < 8; ++r) {
        float v = c[i][j][r] + b;
        size_t idx = (size_t)(rowBase + r) * N + col;
        if (OUT_BF16) ((__bf16*)Cout)[idx] = (__bf16)v;
        else          ((float*)Cout)[idx]  = v;
      }
    }
  }
}

// --------------------------- attention kernel ------------------------------
// One wave per token. 128 threads = 4 tokens / block. 52KB LDS.

__device__ __forceinline__ void unpack8_bf16(uint4 u, float* dst) {
  unsigned w[4] = {u.x, u.y, u.z, u.w};
#pragma unroll
  for (int i = 0; i < 4; ++i) {
    dst[2 * i]     = __uint_as_float(w[i] << 16);
    dst[2 * i + 1] = __uint_as_float(w[i] & 0xffff0000u);
  }
}

__global__ __launch_bounds__(128) void mla_head_attention(const __bf16* __restrict__ qb,
                                                          const __bf16* __restrict__ kvb,
                                                          __bf16* __restrict__ outb) {
  __shared__ float q_s[4][1024];
  __shared__ float k_s[4][1024];
  __shared__ float v_s[4][1024];
  __shared__ float att_s[4][256];

  const int lane = threadIdx.x & 31;
  const int w    = threadIdx.x >> 5;
  const int t    = (int)blockIdx.x * 4 + w;
  const int pos  = t & 4095;            // position within sequence (T = 4096)
  const int hi   = lane >> 4;
  const int nl   = lane & 15;

  // ---- stage q (1024 bf16) and kv (2048 bf16) into LDS as fp32 ----
  const uint4* qrow = (const uint4*)(qb + (size_t)t * 1024);
#pragma unroll
  for (int j = 0; j < 4; ++j) {
    int idx = j * 32 + lane;
    uint4 u = qrow[idx];
    unpack8_bf16(u, &q_s[w][idx * 8]);
  }
  const uint4* kvrow = (const uint4*)(kvb + (size_t)t * 2048);
#pragma unroll
  for (int j = 0; j < 8; ++j) {
    int idx = j * 32 + lane;
    uint4 u = kvrow[idx];
    int e = idx * 8;
    int h = e >> 7;
    int r = e & 127;                    // 8-aligned: stays within k- or v-half
    float tmp[8];
    unpack8_bf16(u, tmp);
    float* dst = (r < 64) ? &k_s[w][h * 64 + r] : &v_s[w][h * 64 + (r - 64)];
#pragma unroll
    for (int q = 0; q < 8; ++q) dst[q] = tmp[q];
  }
  __syncthreads();

  // ---- RoPE on q and k (pairs (i, i+32) within each head) ----
#pragma unroll
  for (int j = 0; j < 16; ++j) {
    int p = j * 32 + lane;              // 16 heads * 32 pairs
    int h = p >> 5;
    int i = p & 31;
    float inv = exp2f(-(float)i * 0.4152410118607702f);   // 10000^(-i/32)
    float ang = (float)pos * inv;
    float s, c;
    __sincosf(ang, &s, &c);
    int d1 = h * 64 + i, d2 = d1 + 32;
    float x1 = q_s[w][d1], x2 = q_s[w][d2];
    q_s[w][d1] = x1 * c - x2 * s;
    q_s[w][d2] = x1 * s + x2 * c;
    float y1 = k_s[w][d1], y2 = k_s[w][d2];
    k_s[w][d1] = y1 * c - y2 * s;
    k_s[w][d2] = y1 * s + y2 * c;
  }
  __syncthreads();

  // ---- scores: att[h][g] = q[h]·k[g] (16x16, K=64) ----
  v8f att = {};
#pragma unroll
  for (int kk = 0; kk < 16; ++kk) {
    int kidx = kk * 4 + hi * 2;         // f32 A/B frag: K = vgpr + 2*hi
    v2f a, b;
    a.x = q_s[w][nl * 64 + kidx];
    a.y = q_s[w][nl * 64 + kidx + 1];
    b.x = k_s[w][nl * 64 + kidx];
    b.y = k_s[w][nl * 64 + kidx + 1];
    att = __builtin_amdgcn_wmma_f32_16x16x4_f32(false, a, false, b,
                                                (short)0, att, false, false);
  }

  // ---- softmax over g (16 lanes per half-wave hold one row) ----
#pragma unroll
  for (int r = 0; r < 8; ++r) {
    float x = att[r] * 0.125f;          // 1/sqrt(64)
    float m = x;
    m = fmaxf(m, __shfl_xor(m, 1));
    m = fmaxf(m, __shfl_xor(m, 2));
    m = fmaxf(m, __shfl_xor(m, 4));
    m = fmaxf(m, __shfl_xor(m, 8));
    float e = __expf(x - m);
    float s = e;
    s += __shfl_xor(s, 1);
    s += __shfl_xor(s, 2);
    s += __shfl_xor(s, 4);
    s += __shfl_xor(s, 8);
    att_s[w][(r + 8 * hi) * 16 + nl] = e / s;
  }
  __syncthreads();

  // ---- out[h][d] = att[h][:] @ v[:][d]  (16x64, K=16) ----
#pragma unroll
  for (int nt = 0; nt < 4; ++nt) {
    v8f o = {};
#pragma unroll
    for (int kk = 0; kk < 4; ++kk) {
      int kidx = kk * 4 + hi * 2;
      v2f a, b;
      a.x = att_s[w][nl * 16 + kidx];
      a.y = att_s[w][nl * 16 + kidx + 1];
      b.x = v_s[w][kidx * 64 + nt * 16 + nl];
      b.y = v_s[w][(kidx + 1) * 64 + nt * 16 + nl];
      o = __builtin_amdgcn_wmma_f32_16x16x4_f32(false, a, false, b,
                                                (short)0, o, false, false);
    }
#pragma unroll
    for (int r = 0; r < 8; ++r) {
      int hrow = r + 8 * hi;
      outb[(size_t)t * 1024 + hrow * 64 + nt * 16 + nl] = (__bf16)o[r];
    }
  }
}

// ------------------------------- launcher ----------------------------------

static inline char* ws_take(char*& p, size_t bytes) {
  char* r = p;
  p += (bytes + 255) & ~(size_t)255;
  return r;
}

extern "C" void kernel_launch(void* const* d_in, const int* in_sizes, int n_in,
                              void* d_out, int out_size, void* d_ws, size_t ws_size,
                              hipStream_t stream) {
  (void)in_sizes; (void)n_in; (void)out_size; (void)ws_size;

  const float* x     = (const float*)d_in[0];
  const float* Wq    = (const float*)d_in[1];
  const float* bq    = (const float*)d_in[2];
  const float* Wdown = (const float*)d_in[3];
  const float* bdown = (const float*)d_in[4];
  const float* Wup   = (const float*)d_in[5];
  const float* bup   = (const float*)d_in[6];
  const float* Wout  = (const float*)d_in[7];
  const float* bout  = (const float*)d_in[8];
  float* out = (float*)d_out;

  constexpr int BT = 4 * 4096;   // 16384 tokens
  constexpr int D  = 1024;
  constexpr int DL = 256;
  constexpr int D2 = 2048;

  char* ws = (char*)d_ws;
  __bf16* xb   = (__bf16*)ws_take(ws, (size_t)BT * D * 2);   // x (bf16)
  __bf16* Wqt  = (__bf16*)ws_take(ws, (size_t)D  * D * 2);   // Wq^T  (N=1024,K=1024)
  __bf16* Wdt  = (__bf16*)ws_take(ws, (size_t)DL * D * 2);   // Wdown^T (N=256,K=1024)
  __bf16* Wut  = (__bf16*)ws_take(ws, (size_t)D2 * DL * 2);  // Wup^T (N=2048,K=256)
  __bf16* Wot  = (__bf16*)ws_take(ws, (size_t)D  * D * 2);   // Wout^T
  __bf16* qbuf = (__bf16*)ws_take(ws, (size_t)BT * D * 2);   // q (bf16)
  __bf16* latb = (__bf16*)ws_take(ws, (size_t)BT * DL * 2);  // kv_latent (bf16)
  __bf16* kvb  = (__bf16*)ws_take(ws, (size_t)BT * D2 * 2);  // kv (bf16)
  __bf16* aob  = (__bf16*)ws_take(ws, (size_t)BT * D * 2);   // attention out (bf16)

  // precision conversions / weight transposes
  cvt_f32_to_bf16<<<(BT * D + 255) / 256, 256, 0, stream>>>(x, xb, BT * D);
  cvt_transpose_bf16<<<(D * D + 255) / 256, 256, 0, stream>>>(Wq, Wqt, D, D);
  cvt_transpose_bf16<<<(D * DL + 255) / 256, 256, 0, stream>>>(Wdown, Wdt, D, DL);
  cvt_transpose_bf16<<<(DL * D2 + 255) / 256, 256, 0, stream>>>(Wup, Wut, DL, D2);
  cvt_transpose_bf16<<<(D * D + 255) / 256, 256, 0, stream>>>(Wout, Wot, D, D);

  // q = x@Wq + bq                         (16384x1024x1024) -> bf16
  gemm_bf16_wmma<true><<<dim3(D / 128, BT / 128), 256, 0, stream>>>(
      xb, Wqt, bq, qbuf, BT, D, D);
  // lat = x@Wdown + bdown                 (16384x256x1024)  -> bf16
  gemm_bf16_wmma<true><<<dim3(DL / 128, BT / 128), 256, 0, stream>>>(
      xb, Wdt, bdown, latb, BT, DL, D);
  // kv = lat@Wup + bup                    (16384x2048x256)  -> bf16
  gemm_bf16_wmma<true><<<dim3(D2 / 128, BT / 128), 256, 0, stream>>>(
      latb, Wut, bup, kvb, BT, D2, DL);

  // per-token RoPE + 16x16 head attention
  mla_head_attention<<<BT / 4, 128, 0, stream>>>(qbuf, kvb, aob);

  // out = att_out@Wout + bout             (16384x1024x1024) -> fp32 d_out
  gemm_bf16_wmma<false><<<dim3(D / 128, BT / 128), 256, 0, stream>>>(
      aob, Wot, bout, out, BT, D, D);
}
